// PointTransformerLayer_56427280335318
// MI455X (gfx1250) — matrix operand order repaired and checked by reading, not compiled, not run on hardware
//
#include <hip/hip_runtime.h>
#include <hip/hip_bf16.h>

typedef _Float16 v16h __attribute__((ext_vector_type(16)));
typedef float    v8f  __attribute__((ext_vector_type(8)));

static constexpr int Bb = 4, Nn = 4096, Kk = 16, D = 64;

__device__ __forceinline__ v8f wmma_f16(v16h a, v16h b, v8f c) {
  // D = A(16x32 f16) * B(32x16 f16) + C(16x16 f32)
  return __builtin_amdgcn_wmma_f32_16x16x32_f16(false, a, false, b, (short)0, c, false, false);
}

// B-tile: pre-swizzled packed weights; lane L holds 16 contiguous halfs.
__device__ __forceinline__ v16h load_btile(const _Float16* __restrict__ wpk, int t, int j, int lane) {
  return *(const v16h*)(wpk + (((t * 4 + j) * 32 + lane) << 4));
}

// A-tile (16x32 f16) from a 64-float row: lane L = row M=L%16.
// slot s<8  -> K = 32t + 8*hi + s      (8 contiguous floats)
// slot s>=8 -> K = 32t + 16 + 8*hi + (s-8)  (8 contiguous floats)
__device__ __forceinline__ v16h build_a(const float* row, int t, int hi) {
  const float* p = row + 32 * t + 8 * hi;
  v16h a;
#pragma unroll
  for (int i = 0; i < 8; ++i) {
    a[i]     = (_Float16)p[i];
    a[i + 8] = (_Float16)p[16 + i];
  }
  return a;
}

// ---------------- weight pre-swizzle (f32 -> f16, B-matrix layout) ----------------
__global__ void pt_pack_w(const float* __restrict__ W, _Float16* __restrict__ dst) {
  int e = blockIdx.x * 256 + threadIdx.x;      // 0..4095
  int s = e & 15;
  int L = (e >> 4) & 31;
  int j = (e >> 9) & 3;
  int t = e >> 11;
  int K   = 32 * t + 8 * (L >> 4) + (s & 7) + ((s >> 3) << 4);      // table order: 8*hi + s, +16 for s >= 8
  int col = 16 * j + (L & 15);
  dst[e] = (_Float16)W[K * 64 + col];
}

// ---------------- KNN: top-16 by (dist, idx), matches top_k(-d, 16) ----------------
__global__ __launch_bounds__(256) void pt_knn(const float* __restrict__ coords,
                                              int* __restrict__ knn) {
  __shared__ float cx[Nn], cy[Nn], cz[Nn], sq[Nn];
  const int b    = blockIdx.x >> 4;
  const int qblk = blockIdx.x & 15;
  const float* cb = coords + (size_t)b * Nn * 3;
  for (int i = threadIdx.x; i < Nn; i += 256) {
    float x = cb[3 * i], y = cb[3 * i + 1], z = cb[3 * i + 2];
    cx[i] = x; cy[i] = y; cz[i] = z;
    sq[i] = x * x + y * y + z * z;
  }
  __syncthreads();

  const int n = qblk * 256 + threadIdx.x;
  const float qx = cx[n], qy = cy[n], qz = cz[n], qs = sq[n];

  float d[16]; int id[16];
#pragma unroll
  for (int j = 0; j < 16; ++j) { d[j] = 3.0e38f; id[j] = 0x7fffffff; }
  float worst = 3.0e38f; int worstj = 0;

  for (int m = 0; m < Nn; ++m) {
    // same expression as reference: sq_n + sq_m - 2*dot
    float dist = qs + sq[m] - 2.f * (qx * cx[m] + qy * cy[m] + qz * cz[m]);
    if (dist < worst) {            // strict <: keeps earliest (lowest) index on ties
#pragma unroll
      for (int j = 0; j < 16; ++j) if (j == worstj) { d[j] = dist; id[j] = m; }
      worst = d[0]; worstj = 0;
#pragma unroll
      for (int j = 1; j < 16; ++j) if (d[j] > worst) { worst = d[j]; worstj = j; }
    }
  }

  // stable ascending selection by (dist, idx)
  unsigned sel = 0;
  int* dst = knn + ((size_t)b * Nn + n) * Kk;
  for (int r = 0; r < 16; ++r) {
    float bd = 3.4e38f; int bid = 0x7fffffff; int bj = 0;
#pragma unroll
    for (int j = 0; j < 16; ++j) {
      bool avail  = ((sel >> j) & 1u) == 0u;
      bool better = avail && ((d[j] < bd) || (d[j] == bd && id[j] < bid));
      if (better) { bd = d[j]; bid = id[j]; bj = j; }
    }
    sel |= (1u << bj);
    dst[r] = bid;
  }
}

// ---------------- stage B: x = feats@W1+b1 ; q/xk/xv = x@{Wq,Wk,Wv} ----------------
__global__ __launch_bounds__(32) void pt_stageB(const float* __restrict__ feats,
                                                const float* __restrict__ b1,
                                                const _Float16* __restrict__ wpk,
                                                float* __restrict__ q,
                                                float* __restrict__ xk,
                                                float* __restrict__ xv) {
  __shared__ float xl[16][64];
  __shared__ float st[16][64];
  const int lane = threadIdx.x;
  const int hi = lane >> 4, lr = lane & 15;
  const int row0 = blockIdx.x * 16;
  const float* frow = feats + (size_t)(row0 + lr) * 64;

  v16h a[2];
#pragma unroll
  for (int t = 0; t < 2; ++t) a[t] = build_a(frow, t, hi);

  v8f cc[4];
#pragma unroll
  for (int j = 0; j < 4; ++j)
#pragma unroll
    for (int r = 0; r < 8; ++r) cc[j][r] = 0.f;
#pragma unroll
  for (int j = 0; j < 4; ++j)
#pragma unroll
    for (int t = 0; t < 2; ++t)
      cc[j] = wmma_f16(a[t], load_btile(wpk /*W1*/, t, j, lane), cc[j]);

#pragma unroll
  for (int j = 0; j < 4; ++j) {
    float bias = b1[16 * j + lr];
#pragma unroll
    for (int r = 0; r < 8; ++r) xl[r + 8 * hi][16 * j + lr] = cc[j][r] + bias;
  }
  __syncthreads();

#pragma unroll
  for (int t = 0; t < 2; ++t) a[t] = build_a(&xl[lr][0], t, hi);

  float* dsts[3] = {q, xk, xv};
#pragma unroll
  for (int w = 0; w < 3; ++w) {
    const _Float16* Wp = wpk + (w + 1) * 4096;   // Wq, Wk, Wv
#pragma unroll
    for (int j = 0; j < 4; ++j)
#pragma unroll
      for (int r = 0; r < 8; ++r) cc[j][r] = 0.f;
#pragma unroll
    for (int j = 0; j < 4; ++j)
#pragma unroll
      for (int t = 0; t < 2; ++t)
        cc[j] = wmma_f16(a[t], load_btile(Wp, t, j, lane), cc[j]);
    __syncthreads();   // previous st readers done
#pragma unroll
    for (int j = 0; j < 4; ++j)
#pragma unroll
      for (int r = 0; r < 8; ++r) st[r + 8 * hi][16 * j + lr] = cc[j][r];
    __syncthreads();
    float* dp = dsts[w] + (size_t)row0 * 64;
    const float* sf = &st[0][0];
#pragma unroll
    for (int i = 0; i < 8; ++i)
      *(float4*)(dp + i * 128 + lane * 4) = *(const float4*)(sf + i * 128 + lane * 4);
  }
}

// ---------------- stage C: per-query attention (one wave = 16 neighbors) ----------------
__global__ __launch_bounds__(32) void pt_stageC(const float* __restrict__ coords,
                                                const float* __restrict__ feats,
                                                const int* __restrict__ knn,
                                                const float* __restrict__ q,
                                                const float* __restrict__ xk,
                                                const float* __restrict__ xv,
                                                const float* __restrict__ Wp1,
                                                const float* __restrict__ bp1,
                                                const float* __restrict__ bp2,
                                                const float* __restrict__ bm1,
                                                const float* __restrict__ bm2,
                                                const float* __restrict__ W2,
                                                const float* __restrict__ b2,
                                                const _Float16* __restrict__ wpk,
                                                float* __restrict__ out,
                                                float* __restrict__ att) {
  __shared__ float rel[16][3];
  __shared__ int   nidx[16];
  __shared__ float pos[16][64];
  __shared__ float vp[16][64];
  __shared__ float tmp[16][64];
  __shared__ float outn[64];

  const int lane = threadIdx.x;
  const int hi = lane >> 4, lr = lane & 15;
  const int qi = blockIdx.x;
  const int b = qi >> 12;          // / 4096
  const int n = qi & 4095;

  if (lane < 16) {
    int m = knn[(size_t)qi * Kk + lane];
    nidx[lane] = m;
#pragma unroll
    for (int c = 0; c < 3; ++c)
      rel[lane][c] = coords[((size_t)b * Nn + n) * 3 + c] - coords[((size_t)b * Nn + m) * 3 + c];
  }
  __syncthreads();

  // t1 = relu(rel @ Wp1 + bp1) built directly into f16 A-layout registers
  const float rx = rel[lr][0], ry = rel[lr][1], rz = rel[lr][2];
  v16h t1a[2];
#pragma unroll
  for (int t = 0; t < 2; ++t) {
#pragma unroll
    for (int s = 0; s < 16; ++s) {
      int f = 32 * t + ((s < 8) ? (8 * hi + s) : (16 + 8 * hi + (s - 8)));
      float v = bp1[f] + rx * Wp1[f] + ry * Wp1[64 + f] + rz * Wp1[128 + f];
      t1a[t][s] = (_Float16)fmaxf(v, 0.f);
    }
  }

  // pos = t1 @ Wp2 + bp2
  const _Float16* Wp2p = wpk + 6 * 4096;
  v8f cc[4];
#pragma unroll
  for (int j = 0; j < 4; ++j)
#pragma unroll
    for (int r = 0; r < 8; ++r) cc[j][r] = 0.f;
#pragma unroll
  for (int j = 0; j < 4; ++j)
#pragma unroll
    for (int t = 0; t < 2; ++t)
      cc[j] = wmma_f16(t1a[t], load_btile(Wp2p, t, j, lane), cc[j]);
#pragma unroll
  for (int j = 0; j < 4; ++j) {
    float bias = bp2[16 * j + lr];
#pragma unroll
    for (int r = 0; r < 8; ++r) pos[r + 8 * hi][16 * j + lr] = cc[j][r] + bias;
  }
  __syncthreads();

  // vp = gather(xv) + pos   (lane owns row lr, feature half hi)
  {
    const float* vr = xv + ((size_t)b * Nn + nidx[lr]) * 64 + 32 * hi;
    float* dp = &vp[lr][32 * hi];
    const float* ps = &pos[lr][32 * hi];
#pragma unroll
    for (int i = 0; i < 32; ++i) dp[i] = vr[i] + ps[i];
  }

  // h = q - k + pos, built into A tiles
  const float* qrow = q + (size_t)qi * 64;
  const float* krow = xk + ((size_t)b * Nn + nidx[lr]) * 64;
  const float* prow = &pos[lr][0];
  v16h ha[2];
#pragma unroll
  for (int t = 0; t < 2; ++t) {
    int base = 32 * t + 8 * hi;
#pragma unroll
    for (int i = 0; i < 8; ++i) {
      ha[t][i]     = (_Float16)(qrow[base + i]      - krow[base + i]      + prow[base + i]);
      ha[t][i + 8] = (_Float16)(qrow[base + 16 + i] - krow[base + 16 + i] + prow[base + 16 + i]);
    }
  }

  // a1 = relu(h @ Wm1 + bm1)
  const _Float16* Wm1p = wpk + 4 * 4096;
#pragma unroll
  for (int j = 0; j < 4; ++j)
#pragma unroll
    for (int r = 0; r < 8; ++r) cc[j][r] = 0.f;
#pragma unroll
  for (int j = 0; j < 4; ++j)
#pragma unroll
    for (int t = 0; t < 2; ++t)
      cc[j] = wmma_f16(ha[t], load_btile(Wm1p, t, j, lane), cc[j]);
#pragma unroll
  for (int j = 0; j < 4; ++j) {
    float bias = bm1[16 * j + lr];
#pragma unroll
    for (int r = 0; r < 8; ++r)
      tmp[r + 8 * hi][16 * j + lr] = fmaxf(cc[j][r] + bias, 0.f);
  }
  __syncthreads();

  v16h aa[2];
#pragma unroll
  for (int t = 0; t < 2; ++t) aa[t] = build_a(&tmp[lr][0], t, hi);

  // logits = a1 @ Wm2 + bm2
  const _Float16* Wm2p = wpk + 5 * 4096;
#pragma unroll
  for (int j = 0; j < 4; ++j)
#pragma unroll
    for (int r = 0; r < 8; ++r) cc[j][r] = 0.f;
#pragma unroll
  for (int j = 0; j < 4; ++j)
#pragma unroll
    for (int t = 0; t < 2; ++t)
      cc[j] = wmma_f16(aa[t], load_btile(Wm2p, t, j, lane), cc[j]);
  __syncthreads();   // done reading tmp (a1); reuse it for attention staging

  // softmax over the 16 neighbors (rows split across lane halves) + weighted sum
#pragma unroll
  for (int j = 0; j < 4; ++j) {
    float vals[8];
    float mx = -3.0e38f;
#pragma unroll
    for (int r = 0; r < 8; ++r) {
      vals[r] = (cc[j][r] + bm2[16 * j + lr]) * 0.125f;   // / sqrt(64)
      mx = fmaxf(mx, vals[r]);
    }
    mx = fmaxf(mx, __shfl_xor(mx, 16));
    float sum = 0.f;
#pragma unroll
    for (int r = 0; r < 8; ++r) { vals[r] = expf(vals[r] - mx); sum += vals[r]; }
    sum += __shfl_xor(sum, 16);
    float inv = 1.f / sum;
    float acc = 0.f;
#pragma unroll
    for (int r = 0; r < 8; ++r) {
      float a = vals[r] * inv;
      int M = r + 8 * hi;
      tmp[M][16 * j + lr] = a;             // attention output staging
      acc += a * vp[M][16 * j + lr];
    }
    acc += __shfl_xor(acc, 16);
    if (hi == 0) outn[16 * j + lr] = acc;
  }
  __syncthreads();

  // attention out: [K=16][D=64] row-major, coalesced float4
  {
    float* ao = att + (size_t)qi * (Kk * D);
    const float* sf = &tmp[0][0];
#pragma unroll
    for (int i = 0; i < 8; ++i)
      *(float4*)(ao + i * 128 + lane * 4) = *(const float4*)(sf + i * 128 + lane * 4);
  }

  // out = outn @ W2 + b2 + feats   (1x64 * 64x64, VALU)
#pragma unroll
  for (int h2 = 0; h2 < 2; ++h2) {
    int f = lane + 32 * h2;
    float acc = b2[f] + feats[(size_t)qi * 64 + f];
    for (int c = 0; c < 64; ++c) acc += outn[c] * W2[c * 64 + f];
    out[(size_t)qi * 64 + f] = acc;
  }
}

extern "C" void kernel_launch(void* const* d_in, const int* in_sizes, int n_in,
                              void* d_out, int out_size, void* d_ws, size_t ws_size,
                              hipStream_t stream) {
  const float* coords = (const float*)d_in[0];
  const float* feats  = (const float*)d_in[1];
  const float* W1  = (const float*)d_in[2];
  const float* b1  = (const float*)d_in[3];
  const float* Wq  = (const float*)d_in[4];
  const float* Wk  = (const float*)d_in[5];
  const float* Wv  = (const float*)d_in[6];
  const float* Wm1 = (const float*)d_in[7];
  const float* bm1 = (const float*)d_in[8];
  const float* Wm2 = (const float*)d_in[9];
  const float* bm2 = (const float*)d_in[10];
  const float* Wp1 = (const float*)d_in[11];
  const float* bp1 = (const float*)d_in[12];
  const float* Wp2 = (const float*)d_in[13];
  const float* bp2 = (const float*)d_in[14];
  const float* W2  = (const float*)d_in[15];
  const float* b2  = (const float*)d_in[16];

  char* ws = (char*)d_ws;
  int*   knnw = (int*)ws;                                        // 1 MiB
  float* q    = (float*)(ws + (1u << 20));                       // 4 MiB
  float* xk   = q  + (size_t)Bb * Nn * D;                        // 4 MiB
  float* xv   = xk + (size_t)Bb * Nn * D;                        // 4 MiB
  _Float16* wpk = (_Float16*)(ws + (1u << 20) + 3ull * Bb * Nn * D * sizeof(float)); // 56 KiB

  // pack order: W1, Wq, Wk, Wv, Wm1, Wm2, Wp2
  const float* srcs[7] = {W1, Wq, Wk, Wv, Wm1, Wm2, Wp2};
  for (int w = 0; w < 7; ++w)
    pt_pack_w<<<16, 256, 0, stream>>>(srcs[w], wpk + (size_t)w * 4096);

  pt_knn<<<Bb * 16, 256, 0, stream>>>(coords, knnw);
  pt_stageB<<<(Bb * Nn) / 16, 32, 0, stream>>>(feats, b1, wpk, q, xk, xv);

  float* outp = (float*)d_out;
  float* attp = outp + (size_t)Bb * Nn * D;
  pt_stageC<<<Bb * Nn, 32, 0, stream>>>(coords, feats, knnw, q, xk, xv,
                                        Wp1, bp1, bp2, bm1, bm2, W2, b2,
                                        wpk, outp, attp);
}
